// SelfAttnLayer_56075093017288
// MI455X (gfx1250) — matrix operand order repaired and checked by reading, not compiled, not run on hardware
//
#include <hip/hip_runtime.h>

// ---------------------------------------------------------------------------
// Fused linear-attention layer for MI455X (gfx1250, wave32, WMMA).
//   out = (attn(relu-split(x @ qkv_w))) @ proj_w + proj_b
// One 128-row group per workgroup; q/k/v/kv live in LDS (~309 KB of 320 KB).
// GEMMs: v_wmma_f32_16x16x32_bf16 (f32 accum).
// GEMM1 B panels staged via global_load_async_to_lds_b128 (ASYNCcnt-tracked,
// double-buffered in LDS; B fragments double-buffered in VGPRs).
// ---------------------------------------------------------------------------

typedef __bf16 bf16;
typedef __attribute__((ext_vector_type(8)))  float v8f;
typedef __attribute__((ext_vector_type(4)))  bf16  v4bf;
typedef __attribute__((ext_vector_type(8)))  bf16  v8bf;
typedef __attribute__((ext_vector_type(16))) bf16  v16bf;

#define NROWS 131072
#define CDIM  256
#define GGRP  128
#define EPSV  0.001f

// LDS strides (bf16 elements); every vector access stays 16B aligned.
#define XSTR 264   // x / y tile row stride  (528 B)
#define TSTR 136   // kT / vT row stride     (272 B)
#define KVSTR 40   // kv tile row stride     (80 B)

static constexpr int OFF_X  = 0;                        // 67584 B (reused as y)
static constexpr int OFF_Q  = OFF_X  + 128 * XSTR * 2;  // 67584
static constexpr int OFF_KT = OFF_Q  + 128 * XSTR * 2;  // 135168
static constexpr int OFF_VT = OFF_KT + 256 * TSTR * 2;  // 204800
static constexpr int OFF_KV = OFF_VT + 256 * TSTR * 2;  // 274432
static constexpr int OFF_S  = OFF_KV + 8 * 32 * KVSTR * 2; // 294912
static constexpr int OFF_Z  = OFF_S  + 256 * 4;         // 295936
static constexpr int OFF_BP = OFF_Z  + 1024 * 4;        // 300032: B-panel dbl buf
static constexpr int BPANEL = 16 * 256 * 2;             // 8192 B per panel
static constexpr int LDS_TOTAL = OFF_BP + 2 * BPANEL;   // 316416 B (< 320 KB)

// ---- CDNA5 async copy: global -> LDS, 16B per lane, ASYNCcnt-tracked ------
// GVS addressing: saddr (uniform 64-bit base, SGPR pair) + vaddr (32-bit
// per-lane byte offset). vdst VGPR carries the LDS byte offset.
__device__ __forceinline__ void async_copy_b128(unsigned lds_off,
                                                const bf16* gbase,
                                                unsigned goff) {
  asm volatile("global_load_async_to_lds_b128 %0, %1, %2 offset:0"
               :
               : "v"(lds_off), "v"(goff), "s"(gbase)
               : "memory");
}
__device__ __forceinline__ void wait_async_le2() {
#if __has_builtin(__builtin_amdgcn_s_wait_asynccnt)
  __builtin_amdgcn_s_wait_asynccnt(2);
#else
  asm volatile("s_wait_asynccnt 0x2" ::: "memory");
#endif
}
__device__ __forceinline__ void wait_async_0() {
#if __has_builtin(__builtin_amdgcn_s_wait_asynccnt)
  __builtin_amdgcn_s_wait_asynccnt(0);
#else
  asm volatile("s_wait_asynccnt 0x0" ::: "memory");
#endif
}

// A operand (M=16,K=32): lane m = row; halves hold K 0..7/16..23, 8..15/24..31.
__device__ __forceinline__ v16bf load_mat_a(const bf16* base, int row_stride,
                                            int row, int kbase, int lane) {
  const int m = lane & 15, half = lane >> 4;
  const bf16* p = base + (size_t)(row + m) * row_stride + kbase + half * 8;
  v8bf lo = *(const v8bf*)(p);
  v8bf hi = *(const v8bf*)(p + 16);
  return __builtin_shufflevector(lo, hi, 0, 1, 2, 3, 4, 5, 6, 7,
                                 8, 9, 10, 11, 12, 13, 14, 15);
}

// B operand (K=32,N=16) from an N-major panel: lane n = col; halves hold K 0..7/16..23, 8..15/24..31.
__device__ __forceinline__ v16bf load_mat_b(const bf16* baseT, int row_stride,
                                            int col, int kbase, int lane) {
  const int n = lane & 15, half = lane >> 4;
  const bf16* p = baseT + (size_t)(col + n) * row_stride + kbase + half * 8;
  v8bf lo = *(const v8bf*)(p);
  v8bf hi = *(const v8bf*)(p + 16);
  return __builtin_shufflevector(lo, hi, 0, 1, 2, 3, 4, 5, 6, 7,
                                 8, 9, 10, 11, 12, 13, 14, 15);
}

__device__ __forceinline__ v8f wmma_bf16(v16bf a, v16bf b, v8f c) {
  return __builtin_amdgcn_wmma_f32_16x16x32_bf16(false, a, false, b,
                                                 (short)0, c, false, false);
}

__device__ __forceinline__ v8bf pack8(v8f a) {
  v8bf r;
#pragma unroll
  for (int i = 0; i < 8; ++i) r[i] = (bf16)a[i];
  return r;
}
__device__ __forceinline__ v8bf pack8_relu(v8f a) {
  v8bf r;
#pragma unroll
  for (int i = 0; i < 8; ++i) r[i] = (bf16)fmaxf(a[i], 0.f);
  return r;
}

// ---------------------------------------------------------------------------
// Prep: f32 weights -> bf16, transposed into N-major 16-column panels.
// ---------------------------------------------------------------------------
__global__ void attn_prep_weights(const float* __restrict__ qkv_w,
                                  const float* __restrict__ proj_w,
                                  bf16* __restrict__ wqkvT,
                                  bf16* __restrict__ wprojT) {
  const int idx = blockIdx.x * 256 + threadIdx.x;      // 0 .. 262143
  if (idx < 48 * 16 * 256) {
    const int k = idx & 255, n = (idx >> 8) & 15, ct = idx >> 12;
    wqkvT[idx] = (bf16)qkv_w[(size_t)k * 768 + ct * 16 + n];
  } else {
    const int r = idx - 48 * 16 * 256;                 // 0 .. 65535
    const int k = r & 255, n = (r >> 8) & 15, ct = r >> 12;
    wprojT[r] = (bf16)proj_w[(size_t)k * 256 + ct * 16 + n];
  }
}

// ---------------------------------------------------------------------------
// Fused layer: one 128-row group per workgroup, 8 waves.
// ---------------------------------------------------------------------------
__global__ void __launch_bounds__(256, 1)
attn_fused(const float* __restrict__ x,
           const bf16*  __restrict__ wqkvT,
           const bf16*  __restrict__ wprojT,
           const float* __restrict__ proj_b,
           float*       __restrict__ out) {
  extern __shared__ char smem[];
  bf16*  xb  = (bf16*)(smem + OFF_X);   // x tile (bf16), later reused as y
  bf16*  qb  = (bf16*)(smem + OFF_Q);   // q[j][c]   row-major, relu'd
  bf16*  ktb = (bf16*)(smem + OFF_KT);  // k^T[c][j] relu'd
  bf16*  vtb = (bf16*)(smem + OFF_VT);  // v^T[c][j]
  bf16*  kvb = (bf16*)(smem + OFF_KV);  // kv per head, d-major: [h][d][c]
  float* sb  = (float*)(smem + OFF_S);  // s[c] = sum_j k[j][c]
  float* zb  = (float*)(smem + OFF_Z);  // z[j*8+h]

  const int g    = blockIdx.x;
  const int tid  = threadIdx.x;
  const int lane = tid & 31;
  const int wave = tid >> 5;
  const int rowbase = wave * 16;        // this wave's 16-row strip (GEMM1/3)
  const int n = lane & 15, half = lane >> 4;

  // Warm the (L2-resident, shared by all WGs) proj panel set.
  __builtin_prefetch(wprojT + tid * 256, 0, 0);

  // ---- Phase 0: stage x tile f32 -> bf16 LDS (packed cvt + b64 stores) ----
  {
    const float4* xg = (const float4*)x + (size_t)g * (GGRP * CDIM / 4);
#pragma unroll 4
    for (int it = 0; it < 32; ++it) {
      const int f4  = tid + it * 256;           // 0..8191
      const int row = f4 >> 6;                  // 64 float4 per row
      const int c4  = (f4 & 63) * 4;
      const float4 v = xg[f4];
      v4bf w = { (bf16)v.x, (bf16)v.y, (bf16)v.z, (bf16)v.w };
      *(v4bf*)(xb + row * XSTR + c4) = w;
    }
  }
  // Kick the first qkv B panel into LDS while x staging drains.
  {
    const unsigned go = tid * 16;               // 256 thr x 32 B = 8 KB panel
    async_copy_b128(OFF_BP + go,        wqkvT, go);
    async_copy_b128(OFF_BP + go + 4096, wqkvT, go + 4096);
  }
  __syncthreads();

  // ---- Phase 1: GEMM1  qkv[128,768] = x @ qkv_w -------------------------
  // A fragments (16 rows x 256 K) live in registers; B panels are DMA'd to
  // LDS once per WG (double-buffered) and shared by all 8 waves.
  v16bf afrag[8];
#pragma unroll
  for (int ks = 0; ks < 8; ++ks)
    afrag[ks] = load_mat_a(xb, XSTR, rowbase, ks * 32, lane);

  for (int ct = 0; ct < 48; ++ct) {
    // Issue DMA for panel ct+1 into the other buffer (safe: that buffer was
    // last read at iteration ct-1, and every iteration ends with a barrier).
    if (ct + 1 < 48) {
      const unsigned go = tid * 16;
      const bf16* wp = wqkvT + (size_t)(ct + 1) * (16 * 256);
      const unsigned dst = OFF_BP + ((ct + 1) & 1) * BPANEL;
      async_copy_b128(dst + go,        wp, go);
      async_copy_b128(dst + go + 4096, wp, go + 4096);
      wait_async_le2();   // panel ct has landed (2 newer ops still in flight)
    } else {
      wait_async_0();
    }
    __syncthreads();      // all waves' panel-ct copies visible

    const bf16* bt = (const bf16*)(smem + OFF_BP + (ct & 1) * BPANEL);
    // Software-pipelined over B fragments: load ks+1 while WMMA ks runs.
    v8f acc = {};
    v16bf bcur = load_mat_b(bt, 256, 0, 0, lane);
#pragma unroll
    for (int ks = 0; ks < 8; ++ks) {
      v16bf bnext;
      if (ks < 7) bnext = load_mat_b(bt, 256, 0, (ks + 1) * 32, lane);
      acc = wmma_bf16(afrag[ks], bcur, acc);
      bcur = bnext;
    }

    if (ct < 16) {        // q: relu, row-major scalar stores
#pragma unroll
      for (int vi = 0; vi < 8; ++vi)
        qb[(rowbase + vi + 8 * half) * XSTR + ct * 16 + n] =
            (bf16)fmaxf(acc[vi], 0.f);
    } else if (ct < 32) { // k: relu, transposed -> one packed b128 store
      *(v8bf*)(ktb + ((ct - 16) * 16 + n) * TSTR + rowbase + 8 * half) =
          pack8_relu(acc);
    } else {              // v: transposed -> one packed b128 store
      *(v8bf*)(vtb + ((ct - 32) * 16 + n) * TSTR + rowbase + 8 * half) =
          pack8(acc);
    }
    __syncthreads();      // reads of buf[ct&1] done before it is re-DMA'd
  }

  // ---- Phase 2: per-head kv = k^T @ v ([32,32], K=128) + s = sum_j k -----
  {
    const int h = wave;                        // one head per wave
#pragma unroll
    for (int mt = 0; mt < 2; ++mt)
#pragma unroll
      for (int nt = 0; nt < 2; ++nt) {
        v8f acc = {};
#pragma unroll
        for (int ks = 0; ks < 4; ++ks) {
          v16bf a = load_mat_a(ktb + (h * 32 + mt * 16) * TSTR, TSTR, 0, ks * 32, lane);
          v16bf b = load_mat_b(vtb + (h * 32 + nt * 16) * TSTR, TSTR, 0, ks * 32, lane);
          acc = wmma_bf16(a, b, acc);
        }
        // D element (c = mt*16 + vi + 8*half, d = nt*16 + n): contiguous c.
        *(v8bf*)(kvb + (h * 32 + nt * 16 + n) * KVSTR + mt * 16 + 8 * half) =
            pack8(acc);
      }
  }
  {  // s[c]: one channel per thread; vector loads over contiguous j.
    const v8bf* kp = (const v8bf*)(ktb + tid * TSTR);
    float acc = 0.f;
#pragma unroll
    for (int j8 = 0; j8 < 16; ++j8) {
      v8bf kv8 = kp[j8];
#pragma unroll
      for (int e = 0; e < 8; ++e) acc += (float)kv8[e];
    }
    sb[tid] = acc;
  }
  __syncthreads();

  // ---- Phase 3: z[j,h] = sum_d s[h,d] * q[j,h,d] -------------------------
#pragma unroll
  for (int it = 0; it < 4; ++it) {
    const int idx = tid + it * 256;            // 0..1023 == j*8 + h
    const int j = idx >> 3, h = idx & 7;
    const v8bf* qp = (const v8bf*)(qb + j * XSTR + h * 32);
    const float* sp = sb + h * 32;
    float acc = 0.f;
#pragma unroll
    for (int c8 = 0; c8 < 4; ++c8) {
      v8bf qv = qp[c8];
#pragma unroll
      for (int e = 0; e < 8; ++e) acc += sp[c8 * 8 + e] * (float)qv[e];
    }
    zb[idx] = acc;
  }
  __syncthreads();

  // ---- Phase 4: y^T = kv^T @ q^T, normalized, bf16 into x's LDS slot -----
  // A = kv tile [d][c] (row-major in LDS), B = q rows (the N-major panel).
  // D element (d = mt*16 + vi + 8*half, j = jt*16 + n): contiguous along d.
  bf16* yb = xb;  // x tile is dead now
  {
    const int jt = wave;                       // one 16-row j-strip per wave
#pragma unroll
    for (int h = 0; h < 8; ++h) {
      const float rz = 1.f / (zb[(jt * 16 + n) * 8 + h] + EPSV);  // lane-fixed j
#pragma unroll
      for (int mt = 0; mt < 2; ++mt) {
        v16bf a = load_mat_a(kvb + (h * 32 + mt * 16) * KVSTR, KVSTR, 0, 0, lane);
        v16bf b = load_mat_b(qb, XSTR, jt * 16, h * 32, lane);
        v8f acc = {};
        acc = wmma_bf16(a, b, acc);
        v8bf r;
#pragma unroll
        for (int vi = 0; vi < 8; ++vi) r[vi] = (bf16)(acc[vi] * rz);
        *(v8bf*)(yb + (jt * 16 + n) * XSTR + h * 32 + mt * 16 + 8 * half) = r;
      }
    }
  }
  __syncthreads();

  // ---- Phase 5: GEMM3  out = y @ proj_w + proj_b -------------------------
  v16bf yfrag[8];
#pragma unroll
  for (int ks = 0; ks < 8; ++ks)
    yfrag[ks] = load_mat_a(yb, XSTR, rowbase, ks * 32, lane);
  for (int ct = 0; ct < 16; ++ct) {
    v8f acc = {};
    const bf16* bt = wprojT + (size_t)ct * (16 * 256);
#pragma unroll
    for (int ks = 0; ks < 8; ++ks)
      acc = wmma_bf16(yfrag[ks], load_mat_b(bt, 256, 0, ks * 32, lane), acc);
    const float pb = proj_b[ct * 16 + n];
#pragma unroll
    for (int vi = 0; vi < 8; ++vi) {
      const int j = rowbase + vi + 8 * half;
      out[(size_t)(g * GGRP + j) * CDIM + ct * 16 + n] = acc[vi] + pb;
    }
  }
}

// ---------------------------------------------------------------------------
extern "C" void kernel_launch(void* const* d_in, const int* in_sizes, int n_in,
                              void* d_out, int out_size, void* d_ws, size_t ws_size,
                              hipStream_t stream) {
  const float* x      = (const float*)d_in[0];
  const float* qkv_w  = (const float*)d_in[1];
  const float* proj_w = (const float*)d_in[2];
  const float* proj_b = (const float*)d_in[3];
  float* out = (float*)d_out;

  bf16* wqkvT  = (bf16*)d_ws;                 // 48*16*256 bf16 = 384 KB
  bf16* wprojT = wqkvT + 48 * 16 * 256;       // 16*16*256 bf16 = 128 KB

  attn_prep_weights<<<1024, 256, 0, stream>>>(qkv_w, proj_w, wqkvT, wprojT);

  (void)hipFuncSetAttribute((const void*)attn_fused,
                            hipFuncAttributeMaxDynamicSharedMemorySize, LDS_TOTAL);
  attn_fused<<<NROWS / GGRP, 256, LDS_TOTAL, stream>>>(x, wqkvT, wprojT, proj_b, out);
}
